// RNN_tanh2_1589137900340
// MI455X (gfx1250) — matrix operand order repaired and checked by reading, not compiled, not run on hardware
//
#include <hip/hip_runtime.h>
#include <cstdint>

// RNN tanh: B=32, T=2048, I=256, H=512
#define BB 32
#define TT 2048
#define II 256
#define HH 512

typedef __attribute__((ext_vector_type(16))) __bf16 v16bf;
typedef __attribute__((ext_vector_type(8)))  float  v8f;

union Frag16 {
    v16bf v;
    __bf16 e[16];
    uint4 q[2];
};

static __device__ __forceinline__ __bf16 f2bf(float f) { return (__bf16)f; }
static __device__ __forceinline__ float  bf2f(__bf16 b) { return (float)b; }

// gfx1250 hardware TANH transcendental (ISA 8.4: TRANS32 ops include TANH).
// v_nops after it respect the TRANS result-use hazard.
static __device__ __forceinline__ float fast_tanh(float x) {
    float r;
    asm volatile("v_tanh_f32 %0, %1\n\tv_nop\n\tv_nop" : "=v"(r) : "v"(x));
    return r;
}

static __device__ __forceinline__ v8f wmma_bf16(v16bf a, v16bf b, v8f c) {
    // D = A(16x32) * B(32x16) + C, f32 accumulate
    return __builtin_amdgcn_wmma_f32_16x16x32_bf16(false, a, false, b,
                                                   (short)0, c, false, false);
}

// ---- Fragment loaders (contiguous k order k = 16*(lane/16) + s, wave32) ----

// A 16x32 bf16 from row-major bf16 [.., lda]: lanes 0-15 rows; lanes 0-15 take
// K 0-15, lanes 16-31 take K 16-31. One contiguous 32B run.
static __device__ __forceinline__ v16bf load_a_bf16(const __bf16* A, int lda,
                                                    int r0, int k0, int lane) {
    int r  = r0 + (lane & 15);
    int kb = k0 + ((lane & 16) ? 16 : 0);
    const __bf16* p = A + (size_t)r * lda + kb;
    Frag16 f;
    f.q[0] = *(const uint4*)(p);
    f.q[1] = *(const uint4*)(p + 8);
    return f.v;
}

// Same A layout but loading from LDS
static __device__ __forceinline__ v16bf load_a_lds(const __bf16* A, int lda,
                                                   int r0, int k0, int lane) {
    int r  = r0 + (lane & 15);
    int kb = k0 + ((lane & 16) ? 16 : 0);
    const __bf16* p = A + r * lda + kb;
    Frag16 f;
    f.q[0] = *(const uint4*)(p);
    f.q[1] = *(const uint4*)(p + 8);
    return f.v;
}

// B 32x16 bf16 from bf16 [N,K] row-major. Lane = col n; lanes 16-31 take K+16;
// each lane reads 16 contiguous K values (32B = two b128 loads).
static __device__ __forceinline__ v16bf load_b_bf16(const __bf16* Bm, int ldk,
                                                    int n0, int k0, int lane) {
    int n  = n0 + (lane & 15);
    int kb = k0 + ((lane & 16) ? 16 : 0);
    const __bf16* p = Bm + (size_t)n * ldk + kb;
    Frag16 f;
    f.q[0] = *(const uint4*)(p);
    f.q[1] = *(const uint4*)(p + 8);
    return f.v;
}

// B 32x16 from f32 [N,K] row-major with conversion (one-time W preload only)
static __device__ __forceinline__ v16bf load_b_f32(const float* Bm, int ldk,
                                                   int n0, int k0, int lane) {
    int n  = n0 + (lane & 15);
    int kb = k0 + ((lane & 16) ? 16 : 0);
    const float* p = Bm + (size_t)n * ldk + kb;
    Frag16 f;
#pragma unroll
    for (int e = 0; e < 16; ++e) f.e[e] = f2bf(p[e]);
    return f.v;
}

// ---------------- Kernel 0: f32 -> bf16 conversion (x, u_w, o_w) ------------
__global__ __launch_bounds__(256) void k_cvt(const float* __restrict__ src,
                                             __bf16* __restrict__ dst, int n4) {
    int i = blockIdx.x * 256 + threadIdx.x;
    if (i < n4) {
        float4 v = ((const float4*)src)[i];
        __bf16 o[4] = {f2bf(v.x), f2bf(v.y), f2bf(v.z), f2bf(v.w)};
        ((uint2*)dst)[i] = *(const uint2*)o;
    }
}

// ---------------- Kernel 1: xu = x @ u_w^T + u_b  (bf16 in/out) -------------
// grid (H/16, B*T/16/8), block 256 (8 waves, 1 tile each)
__global__ __launch_bounds__(256) void k_inproj(const __bf16* __restrict__ xb,
                                                const __bf16* __restrict__ u_wb,
                                                const float* __restrict__ u_b,
                                                __bf16* __restrict__ xu) {
    int lane = threadIdx.x & 31;
    int wave = threadIdx.x >> 5;
    int m0 = (blockIdx.y * 8 + wave) * 16;
    int n0 = blockIdx.x * 16;
    int n  = n0 + (lane & 15);
    float bias = u_b[n];
    v8f acc;
#pragma unroll
    for (int v = 0; v < 8; ++v) acc[v] = bias;
#pragma unroll
    for (int kk = 0; kk < II; kk += 32) {
        v16bf a = load_a_bf16(xb, II, m0, kk, lane);
        v16bf b = load_b_bf16(u_wb, II, n0, kk, lane);
        acc = wmma_bf16(a, b, acc);
    }
    int rbase = (lane & 16) ? 8 : 0;
#pragma unroll
    for (int v = 0; v < 8; ++v) {
        int r = m0 + rbase + v;
        xu[(size_t)r * HH + n] = f2bf(acc[v]);
    }
}

// ---------------- Kernel 2: persistent recurrence -----------------
// Single workgroup (1 WGP), 32 waves. W (512x512 bf16) split residency:
//   f=0..7  (K 0..255)  pinned in VGPRs (64 VGPRs/lane)
//   f=8..15 (K 256..511) in LDS, wave-shaped lane-direct layout (256 KB)
// Hidden state (33 KB) also in LDS -> 295 KB static LDS (WGP has 320 KB).
// xu loads: registers pipeline t+1, global_prefetch_b8 touches t+2.
// Pipeline indices are clamped (not branched) so the hot loop has one body.
__global__ __launch_bounds__(1024, 1) void k_recur(const __bf16* __restrict__ xu,
                                                   const float* __restrict__ w_w,
                                                   const float* __restrict__ w_b,
                                                   __bf16* __restrict__ hs) {
    constexpr int LDH = HH + 8;                  // row stride 1040B -> 4-bank skew
    __shared__ __align__(16) __bf16 h_lds[BB][LDH];   // 33,280 B
    __shared__ uint4 w_lds[32][8][32][2];             // [wave][f-8][lane][32B] = 262,144 B

    int lane = threadIdx.x & 31;
    int wave = threadIdx.x >> 5;
    int n0 = wave * 16;
    int n  = n0 + (lane & 15);

    // One-time: pin K=0..255 slice in registers, park K=256..511 in LDS
    v16bf wfrag[8];
#pragma unroll
    for (int f = 0; f < 8; ++f)
        wfrag[f] = load_b_f32(w_w, HH, n0, f * 32, lane);
#pragma unroll
    for (int f = 8; f < 16; ++f) {
        Frag16 fr;
        fr.v = load_b_f32(w_w, HH, n0, f * 32, lane);
        w_lds[wave][f - 8][lane][0] = fr.q[0];   // wave-private: no barrier needed
        w_lds[wave][f - 8][lane][1] = fr.q[1];
    }
    float wb = w_b[n];

    // h0 = 0
    for (int i = threadIdx.x; i < BB * LDH; i += 1024)
        (&h_lds[0][0])[i] = __builtin_bit_cast(__bf16, (uint16_t)0);
    __syncthreads();

    int rbase = (lane & 16) ? 8 : 0;

    // pipeline: xu values for the *current* step live in xun0/xun1
    float xun0[8], xun1[8];
#pragma unroll
    for (int v = 0; v < 8; ++v) {
        xun0[v] = bf2f(xu[((size_t)(rbase + v) * TT) * HH + n]);
        xun1[v] = bf2f(xu[((size_t)(16 + rbase + v) * TT) * HH + n]);
    }

    for (int t = 0; t < TT; ++t) {
        int tn = (t + 1 < TT) ? t + 1 : TT - 1;  // clamp: uniform loop body
        int tp = (t + 2 < TT) ? t + 2 : TT - 1;

        // C init = xu(:, t, n) + w_b[n]
        v8f acc0, acc1;
#pragma unroll
        for (int v = 0; v < 8; ++v) {
            acc0[v] = xun0[v] + wb;
            acc1[v] = xun1[v] + wb;
        }
        // touch step t+2's lines (lane = batch index b: one global_prefetch_b8
        // covers all 32 lines this wave will need at t+2)
        {
            const __bf16* pf = xu + ((size_t)lane * TT + tp) * HH + n0;
            __builtin_prefetch((const void*)pf, 0, 1);
        }
        // reload registers for step t+1 (overlaps the WMMA chain below)
#pragma unroll
        for (int v = 0; v < 8; ++v) {
            xun0[v] = bf2f(xu[((size_t)(rbase + v) * TT + tn) * HH + n]);
            xun1[v] = bf2f(xu[((size_t)(16 + rbase + v) * TT + tn) * HH + n]);
        }
        // acc += h(32x512) @ W^T(:, n0:n0+16): register half then LDS half
#pragma unroll
        for (int f = 0; f < 8; ++f) {
            v16bf a0 = load_a_lds(&h_lds[0][0], LDH, 0,  f * 32, lane);
            v16bf a1 = load_a_lds(&h_lds[0][0], LDH, 16, f * 32, lane);
            acc0 = wmma_bf16(a0, wfrag[f], acc0);
            acc1 = wmma_bf16(a1, wfrag[f], acc1);
        }
#pragma unroll
        for (int f = 8; f < 16; ++f) {
            Frag16 wv;
            wv.q[0] = w_lds[wave][f - 8][lane][0];
            wv.q[1] = w_lds[wave][f - 8][lane][1];
            v16bf a0 = load_a_lds(&h_lds[0][0], LDH, 0,  f * 32, lane);
            v16bf a1 = load_a_lds(&h_lds[0][0], LDH, 16, f * 32, lane);
            acc0 = wmma_bf16(a0, wv.v, acc0);
            acc1 = wmma_bf16(a1, wv.v, acc1);
        }
        __syncthreads();                         // all reads of h done
#pragma unroll
        for (int v = 0; v < 8; ++v) {
            int b0 = rbase + v;
            int b1 = 16 + rbase + v;
            __bf16 h0 = f2bf(fast_tanh(acc0[v]));
            __bf16 h1 = f2bf(fast_tanh(acc1[v]));
            h_lds[b0][n] = h0;
            h_lds[b1][n] = h1;
            hs[((size_t)b0 * TT + t) * HH + n] = h0;
            hs[((size_t)b1 * TT + t) * HH + n] = h1;
        }
        __syncthreads();                         // new h visible
    }
}

// ---------------- Kernel 3: out = sigmoid(hs @ o_w^T + o_b) ----------------
// grid (I/16, B*T/16/8), block 256
__global__ __launch_bounds__(256) void k_outproj(const __bf16* __restrict__ hs,
                                                 const __bf16* __restrict__ o_wb,
                                                 const float* __restrict__ o_b,
                                                 float* __restrict__ out) {
    int lane = threadIdx.x & 31;
    int wave = threadIdx.x >> 5;
    int m0 = (blockIdx.y * 8 + wave) * 16;
    int n0 = blockIdx.x * 16;
    int n  = n0 + (lane & 15);
    float bias = o_b[n];
    v8f acc;
#pragma unroll
    for (int v = 0; v < 8; ++v) acc[v] = bias;
#pragma unroll
    for (int kk = 0; kk < HH; kk += 32) {
        v16bf a = load_a_bf16(hs, HH, m0, kk, lane);
        v16bf b = load_b_bf16(o_wb, HH, n0, kk, lane);
        acc = wmma_bf16(a, b, acc);
    }
    int rbase = (lane & 16) ? 8 : 0;
#pragma unroll
    for (int v = 0; v < 8; ++v) {
        int r = m0 + rbase + v;
        float s = 1.0f / (1.0f + __expf(-acc[v]));
        out[(size_t)r * II + n] = s;
    }
}

extern "C" void kernel_launch(void* const* d_in, const int* in_sizes, int n_in,
                              void* d_out, int out_size, void* d_ws, size_t ws_size,
                              hipStream_t stream) {
    const float* x   = (const float*)d_in[0];
    const float* u_w = (const float*)d_in[1];
    const float* u_b = (const float*)d_in[2];
    const float* w_w = (const float*)d_in[3];
    const float* w_b = (const float*)d_in[4];
    const float* o_w = (const float*)d_in[5];
    const float* o_b = (const float*)d_in[6];
    float* out = (float*)d_out;

    // workspace layout (bf16): xu[B*T*H] | hs[B*T*H] | x[B*T*I] | u_w | o_w
    __bf16* xu   = (__bf16*)d_ws;
    __bf16* hs   = xu + (size_t)BB * TT * HH;
    __bf16* xb   = hs + (size_t)BB * TT * HH;
    __bf16* u_wb = xb + (size_t)BB * TT * II;
    __bf16* o_wb = u_wb + (size_t)HH * II;

    // f32 -> bf16 pre-conversions
    {
        int n4 = (BB * TT * II) / 4;
        k_cvt<<<dim3((n4 + 255) / 256), dim3(256), 0, stream>>>(x, xb, n4);
        int w4 = (HH * II) / 4;
        k_cvt<<<dim3((w4 + 255) / 256), dim3(256), 0, stream>>>(u_w, u_wb, w4);
        k_cvt<<<dim3((w4 + 255) / 256), dim3(256), 0, stream>>>(o_w, o_wb, w4);
    }

    dim3 blk(256);
    dim3 g1(HH / 16, (BB * TT) / (16 * 8));
    k_inproj<<<g1, blk, 0, stream>>>(xb, u_wb, u_b, xu);

    k_recur<<<dim3(1), dim3(1024), 0, stream>>>(xu, w_w, w_b, hs);

    dim3 g3(II / 16, (BB * TT) / (16 * 8));
    k_outproj<<<g3, blk, 0, stream>>>(hs, o_wb, o_b, out);
}
